// TruncatedHistoryAttn_11149735100756
// MI455X (gfx1250) — matrix operand order repaired and checked by reading, not compiled, not run on hardware
//
#include <hip/hip_runtime.h>

typedef unsigned short u16;
typedef __attribute__((ext_vector_type(16))) __bf16 v16bf;
typedef __attribute__((ext_vector_type(8)))  float  v8f;

#define D 512
#define S 512
#define B 32
#define NH 5

union Frag {
    v16bf v;
    uint4 q[2];
};

// single-instruction packed f32->bf16 (RNE), CDNA VALU converter
__device__ __forceinline__ unsigned cvt_pk_bf16(float lo, float hi) {
    unsigned r;
    asm("v_cvt_pk_bf16_f32 %0, %1, %2" : "=v"(r) : "v"(lo), "v"(hi));
    return r;
}
__device__ __forceinline__ u16 f32_to_bf16(float f) {
    return (u16)cvt_pk_bf16(f, 0.f);
}

// native transcendental tanh (single trans-pipe op, co-executes with VALU)
__device__ __forceinline__ float tanh_hw(float x) {
    float r;
    asm("v_tanh_f32 %0, %1" : "=v"(r) : "v"(x));
    return r;
}

// CDNA5 async global->LDS copy (ASYNCcnt-tracked), 8B per lane
__device__ __forceinline__ void async_copy_b64(void* lds_dst, const void* gsrc) {
    unsigned lds = (unsigned)(size_t)lds_dst;           // addr[31:0] = LDS offset
    unsigned long long ga = (unsigned long long)(size_t)gsrc;
    asm volatile("global_load_async_to_lds_b64 %0, %1, off"
                 :: "v"(lds), "v"(ga) : "memory");
}
__device__ __forceinline__ void wait_async0() {
    asm volatile("s_wait_asynccnt 0" ::: "memory");
}

__device__ __forceinline__ v8f wmma_bf16(const Frag& a, const Frag& b, v8f c) {
    return __builtin_amdgcn_wmma_f32_16x16x32_bf16(
        false, a.v, false, b.v, (short)0, c, false, false);
}

// ---------------------------------------------------------------------------
// K0a: WT[n][k] = bf16(W[k][n])   (transposed bf16 weights, B-operand layout)
// ---------------------------------------------------------------------------
__global__ void prep_wt(const float* __restrict__ W, u16* __restrict__ WT) {
    int idx = blockIdx.x * blockDim.x + threadIdx.x;   // D*D threads
    int n = idx >> 9, k = idx & (D - 1);
    WT[(size_t)n * D + k] = f32_to_bf16(W[(size_t)k * D + n]);
}

// K0b: Hbf = bf16(H), packed converts, 2 elems/thread
__global__ void prep_hbf(const float* __restrict__ H, u16* __restrict__ Hbf) {
    size_t idx = ((size_t)blockIdx.x * blockDim.x + threadIdx.x) * 2;
    float2 hv = *(const float2*)(H + idx);
    *(unsigned*)(Hbf + idx) = cvt_pk_bf16(hv.x, hv.y);
}

// ---------------------------------------------------------------------------
// K1: Aout[16384,512] = Hbf @ W  (bf16 WMMA, fp32 accum)
// block = 256 thr (8 waves); tile = 32(M) x 128(N); async LDS staging of A
// ---------------------------------------------------------------------------
__global__ void __launch_bounds__(256) precompute_gemm(
    const u16* __restrict__ Hbf, const u16* __restrict__ WT,
    float* __restrict__ Aout) {
    __shared__ u16 As[32 * 32];           // bf16 A tile, row stride 32
    const int tid = threadIdx.x, lane = tid & 31, wave = tid >> 5;
    const int m0 = blockIdx.x * 32;
    const int ntile = blockIdx.y * 128 + wave * 16;

    v8f acc0 = {0.f,0.f,0.f,0.f,0.f,0.f,0.f,0.f};
    v8f acc1 = {0.f,0.f,0.f,0.f,0.f,0.f,0.f,0.f};

    const int m = lane & 15, khalf = lane >> 4;     // A swizzle (ISA 7.12.2)
    const int nB = lane & 15, ksel = lane >> 4;     // B swizzle
    const int sr = tid >> 3, seg = (tid & 7) * 4;   // staging: 8B per thread

    for (int k0 = 0; k0 < D; k0 += 32) {
        // async global->LDS: 32x32 bf16 tile (2KB), 8B per thread
        async_copy_b64(As + sr * 32 + seg,
                       Hbf + (size_t)(m0 + sr) * D + k0 + seg);
        wait_async0();
        __syncthreads();

        Frag b;
        {
            const uint4* bp = (const uint4*)(WT + (size_t)(ntile + nB) * D + k0 + ksel * 8);
            b.q[0] = bp[0]; b.q[1] = bp[2];
        }
        Frag a0, a1;
        {
            const uint4* ap0 = (const uint4*)(As + m * 32 + khalf * 8);
            a0.q[0] = ap0[0]; a0.q[1] = ap0[2];          // K runs {kh*8..}, {16+kh*8..}
            const uint4* ap1 = (const uint4*)(As + (16 + m) * 32 + khalf * 8);
            a1.q[0] = ap1[0]; a1.q[1] = ap1[2];
        }
        acc0 = wmma_bf16(a0, b, acc0);
        acc1 = wmma_bf16(a1, b, acc1);
        __syncthreads();
    }

    const int n = lane & 15, hi = (lane >> 4) * 8;
    for (int r = 0; r < 8; ++r) {
        Aout[(size_t)(m0 + hi + r) * D + ntile + n]      = acc0[r];
        Aout[(size_t)(m0 + 16 + hi + r) * D + ntile + n] = acc1[r];
    }
}

// ---------------------------------------------------------------------------
// K2: persistent sequential scan. 2 blocks x 512 thr (16 waves); block owns 16
// batch rows (one WMMA M-tile). Per step: tanh/dot -> softmax -> fused
// elementwise -> WMMA GEMM (h~ @ W3) into 5-slot ring.
// ---------------------------------------------------------------------------
#define ASTRIDE 520   // LDS row pad: 1040B = 65*16 -> 16B aligned, no bank clash

__global__ void __launch_bounds__(512, 1) seq_scan(
    const float* __restrict__ H, const float* __restrict__ v,
    const u16* __restrict__ W3T, const float* __restrict__ A1,
    const float* __restrict__ A2, float* __restrict__ g3ring,
    float* __restrict__ out) {
    __shared__ float v_s[D];
    __shared__ float logits[16][NH];
    __shared__ float wts[16][NH];
    __shared__ u16 Abuf[16 * ASTRIDE];    // h~ in bf16, WMMA A staging

    const int tid = threadIdx.x, lane = tid & 31, wave = tid >> 5;
    const int bm = blockIdx.x * 16;

    if (tid < D) v_s[tid] = v[tid];
    __syncthreads();

    for (int t = 0; t < S; ++t) {
        // ---- phase A: logits[row][k] = sum_d tanh(A1+A2+G3) * v  (wave = row)
        // lane owns d in [lane*16, lane*16+16) -> all streams are b128 loads
        {
            const int b = bm + wave;
            const int dbase = lane * 16;
            float a2r[16], vr[16];
            const float* a2p = A2 + ((size_t)b * S + t) * D + dbase;
            #pragma unroll
            for (int ii = 0; ii < 4; ++ii) {
                float4 x = *(const float4*)(a2p + ii * 4);
                float4 y = *(const float4*)(v_s + dbase + ii * 4);
                a2r[ii*4+0]=x.x; a2r[ii*4+1]=x.y; a2r[ii*4+2]=x.z; a2r[ii*4+3]=x.w;
                vr [ii*4+0]=y.x; vr [ii*4+1]=y.y; vr [ii*4+2]=y.z; vr [ii*4+3]=y.w;
            }
            for (int k = 0; k < NH; ++k) {
                const int tp = t - NH + k;
                float part = 0.f;
                if (tp >= 0) {
                    const float* a1p = A1 + ((size_t)b * S + tp) * D + dbase;
                    const float* g3p = g3ring + ((size_t)(tp % NH) * B + b) * D + dbase;
                    #pragma unroll
                    for (int ii = 0; ii < 4; ++ii) {
                        float4 xa = *(const float4*)(a1p + ii * 4);
                        float4 xg = *(const float4*)(g3p + ii * 4);
                        part += tanh_hw(xa.x + a2r[ii*4+0] + xg.x) * vr[ii*4+0];
                        part += tanh_hw(xa.y + a2r[ii*4+1] + xg.y) * vr[ii*4+1];
                        part += tanh_hw(xa.z + a2r[ii*4+2] + xg.z) * vr[ii*4+2];
                        part += tanh_hw(xa.w + a2r[ii*4+3] + xg.w) * vr[ii*4+3];
                    }
                } else {        // zero-padded history rows: temp = tanh(h_t W2)
                    #pragma unroll
                    for (int i = 0; i < 16; ++i) part += tanh_hw(a2r[i]) * vr[i];
                }
                for (int off = 16; off > 0; off >>= 1)
                    part += __shfl_down(part, off, 32);
                if (lane == 0) logits[wave][k] = part;
            }
        }
        __syncthreads();

        // ---- phase B: softmax over history axis (5), one lane per row
        if (wave == 0 && lane < 16) {
            float mx = logits[lane][0];
            for (int k = 1; k < NH; ++k) mx = fmaxf(mx, logits[lane][k]);
            float e[NH], s = 0.f;
            for (int k = 0; k < NH; ++k) { e[k] = __expf(logits[lane][k] - mx); s += e[k]; }
            float inv = 1.f / s;
            for (int k = 0; k < NH; ++k) wts[lane][k] = e[k] * inv;
        }
        __syncthreads();

        // ---- phase C: h~ = h_t + relu(sum_k w_k * h~_{t-5+k}); write out + Abuf
        {
            const int d = tid;   // coalesced across lanes
            #pragma unroll
            for (int r = 0; r < 16; ++r) {
                const int b = bm + r;
                float hh = 0.f;
                #pragma unroll
                for (int k = 0; k < NH; ++k) {
                    int tp = t - NH + k;
                    if (tp >= 0) hh += wts[r][k] * out[((size_t)b * S + tp) * D + d];
                }
                float res = H[((size_t)b * S + t) * D + d] + fmaxf(hh, 0.f);
                out[((size_t)b * S + t) * D + d] = res;
                Abuf[r * ASTRIDE + d] = f32_to_bf16(res);
            }
        }
        __threadfence_block();
        __syncthreads();

        // ---- phase D: G3_new = h~ @ W3  ([16,512]x[512,512], WMMA)
        {
            const int m = lane & 15, khalf = lane >> 4;
            const int nB = lane & 15, ksel = lane >> 4;
            const int nt0 = wave, nt1 = wave + 16;      // two N tiles per wave
            v8f acc0 = {0.f,0.f,0.f,0.f,0.f,0.f,0.f,0.f};
            v8f acc1 = {0.f,0.f,0.f,0.f,0.f,0.f,0.f,0.f};
            for (int k0 = 0; k0 < D; k0 += 32) {
                Frag a;
                const uint4* ap = (const uint4*)(Abuf + m * ASTRIDE + k0 + khalf * 8);
                a.q[0] = ap[0]; a.q[1] = ap[2];
                Frag b0, b1;
                const uint4* bp0 = (const uint4*)(W3T + (size_t)(nt0 * 16 + nB) * D + k0 + ksel * 8);
                b0.q[0] = bp0[0]; b0.q[1] = bp0[2];
                const uint4* bp1 = (const uint4*)(W3T + (size_t)(nt1 * 16 + nB) * D + k0 + ksel * 8);
                b1.q[0] = bp1[0]; b1.q[1] = bp1[2];
                if (k0 + 32 < D)   // keep next W3^T k-slab warm (global_prefetch_b8)
                    __builtin_prefetch(W3T + (size_t)(nt0 * 16 + nB) * D + k0 + 32, 0, 1);
                acc0 = wmma_bf16(a, b0, acc0);
                acc1 = wmma_bf16(a, b1, acc1);
            }
            float* g3o = g3ring + (size_t)(t % NH) * B * D;
            const int n = lane & 15, hi = (lane >> 4) * 8;
            #pragma unroll
            for (int r = 0; r < 8; ++r) {
                g3o[(size_t)(bm + hi + r) * D + nt0 * 16 + n] = acc0[r];
                g3o[(size_t)(bm + hi + r) * D + nt1 * 16 + n] = acc1[r];
            }
        }
        __threadfence_block();
        __syncthreads();
    }
}

// ---------------------------------------------------------------------------
extern "C" void kernel_launch(void* const* d_in, const int* in_sizes, int n_in,
                              void* d_out, int out_size, void* d_ws, size_t ws_size,
                              hipStream_t stream) {
    const float* H  = (const float*)d_in[0];   // [32,512,512]
    const float* v  = (const float*)d_in[1];   // [512,1]
    const float* W1 = (const float*)d_in[2];   // [512,512]
    const float* W2 = (const float*)d_in[3];
    const float* W3 = (const float*)d_in[4];
    float* out = (float*)d_out;

    char* ws = (char*)d_ws;
    u16* W1T = (u16*)ws;                               // 512 KB each
    u16* W2T = W1T + (size_t)D * D;
    u16* W3T = W2T + (size_t)D * D;
    u16* Hbf = W3T + (size_t)D * D;                    // 16.8 MB
    float* A1 = (float*)(Hbf + (size_t)B * S * D);     // 33.5 MB each
    float* A2 = A1 + (size_t)B * S * D;
    float* g3ring = A2 + (size_t)B * S * D;            // 5*32*512 f32

    // K0: bf16 transposed weights + bf16 H
    prep_wt<<<dim3((D * D) / 256), 256, 0, stream>>>(W1, W1T);
    prep_wt<<<dim3((D * D) / 256), 256, 0, stream>>>(W2, W2T);
    prep_wt<<<dim3((D * D) / 256), 256, 0, stream>>>(W3, W3T);
    prep_hbf<<<dim3((B * S * D) / 512), 256, 0, stream>>>(H, Hbf);

    // K1: A1 = H@W1, A2 = H@W2 (parallel, WMMA + async LDS staging)
    precompute_gemm<<<dim3((B * S) / 32, D / 128), 256, 0, stream>>>(Hbf, W1T, A1);
    precompute_gemm<<<dim3((B * S) / 32, D / 128), 256, 0, stream>>>(Hbf, W2T, A2);

    // K2: persistent scan, one block per 16 batch rows (independent recurrences)
    seq_scan<<<dim3(B / 16), 512, 0, stream>>>(H, v, W3T, A1, A2, g3ring, out);
}
